// RoutingCaps1_78245714198906
// MI455X (gfx1250) — matrix operand order repaired and checked by reading, not compiled, not run on hardware
//
#include <hip/hip_runtime.h>
#include <hip/hip_bf16.h>

typedef __bf16 bf16;
typedef __attribute__((ext_vector_type(16))) __bf16 v16bf;
typedef __attribute__((ext_vector_type(8)))  __bf16 v8bf;
typedef __attribute__((ext_vector_type(4)))  __bf16 v4bf;
typedef __attribute__((ext_vector_type(8)))  float  v8f;

#define B_     128
#define N0_    64
#define D0_    32
#define D1_    32
#define N1_    64
#define NI_    2048       // N1*D1
#define KCONV  65
#define KP_    2080       // D1*KCONV
#define EPS_Z_ 1.1920929e-7f

__device__ __forceinline__ int lane_id() { return threadIdx.x & 31; }

// ---- WMMA fragment helpers per CDNA5 ISA 7.12.2 (wave32) ----
__device__ __forceinline__ v16bf ld_frag2(const bf16* p0, const bf16* p1) {
  v8bf lo = *(const v8bf*)p0;
  v8bf hi = *(const v8bf*)p1;
  return __builtin_shufflevector(lo, hi, 0, 1, 2, 3, 4, 5, 6, 7,
                                 8, 9, 10, 11, 12, 13, 14, 15);
}
// A (bf16, 16x32) from row-major src p[m*ld + k]: two contiguous 8-elem runs.
__device__ __forceinline__ v16bf load_A_rm(const bf16* p, int ld) {
  const int l = lane_id(), m = l & 15, hi = l >> 4;
  const bf16* r = p + m * ld + (hi << 3);
  return ld_frag2(r, r + 16);
}
// Fragment from a lane-swizzled tile: 16 contiguous bf16 per lane.
__device__ __forceinline__ v16bf load_frag_sw(const bf16* tile) {
  const bf16* r = tile + lane_id() * 16;
  return ld_frag2(r, r + 8);
}
// Swizzled B element (k in 0..31, n in 0..15): lane=n+((k>>3)&1)*16,
// slot=(k>>4)*8+(k&7).
__device__ __forceinline__ int b_sw_idx(int k, int n) {
  return (((((k >> 3) & 1) << 4) | n) << 4) | (((k >> 4) << 3) | (k & 7));
}
// Swizzled A element (m in 0..15, k in 0..31): lane=m+((k>>3)&1)*16,
// slot=(k>>4)*8+(k&7).
__device__ __forceinline__ int a_sw_idx(int m, int k) {
  return ((m + (((k >> 3) & 1) << 4)) << 4) | (((k >> 4) << 3) | (k & 7));
}
__device__ __forceinline__ v8f wmma_bf16(v16bf a, v16bf b, v8f c) {
  return __builtin_amdgcn_wmma_f32_16x16x32_bf16(false, a, false, b, (short)0, c,
                                                 false, false);
}

// ---- K0: f32 -> bf16 convert, 8 elements/thread ----
__global__ void k0_cvt8(const float* __restrict__ w, bf16* __restrict__ o, int n8) {
  int i = blockIdx.x * 256 + threadIdx.x;
  if (i < n8) {
    float4 a = ((const float4*)w)[2 * i];
    float4 b = ((const float4*)w)[2 * i + 1];
    v8bf v = {(bf16)a.x, (bf16)a.y, (bf16)a.z, (bf16)a.w,
              (bf16)b.x, (bf16)b.y, (bf16)b.z, (bf16)b.w};
    ((v8bf*)o)[i] = v;
  }
}

// ---- K1: u[b,k,z,j] = sum_i x[b,j,i] * W[k,j,i,z]  (per (j,k) GEMM 128x32x32) ----
__global__ void k1_transform(const float* __restrict__ x, const float* __restrict__ W,
                             float* __restrict__ utmp) {
  const int j = blockIdx.x;   // N0
  const int k = blockIdx.y;   // N1
  __shared__ alignas(16) bf16 xs[128 * 32];     // row-major [b][i]
  __shared__ alignas(16) bf16 wss[2 * 512];     // 2 swizzled B tiles
  const int tid = threadIdx.x;                  // 128 threads = 4 waves
  for (int e = tid; e < 1024; e += 128) {       // 4-f32 chunks
    int b = e >> 3, i0 = (e & 7) << 2;
    float4 v = *(const float4*)(x + b * (N0_ * D0_) + j * D0_ + i0);
    v4bf o = {(bf16)v.x, (bf16)v.y, (bf16)v.z, (bf16)v.w};
    *(v4bf*)(xs + e * 4) = o;
  }
  for (int e = tid; e < 256; e += 128) {        // 4-f32 chunks of W tile
    int i = e >> 3, z0 = (e & 7) << 2;          // i = K dim, z0 = N base
    float4 v = *(const float4*)(W + (size_t)(k * N0_ + j) * (D0_ * D1_) + e * 4);
    float vv[4] = {v.x, v.y, v.z, v.w};
    int tile = z0 >> 4;                          // chunk stays in one 16-block
#pragma unroll
    for (int q = 0; q < 4; ++q)
      wss[(tile << 9) + b_sw_idx(i, (z0 + q) & 15)] = (bf16)vv[q];
  }
  __syncthreads();
  const int w = tid >> 5, l = tid & 31, n16 = l & 15, hi = l >> 4;
#pragma unroll
  for (int mt = 0; mt < 2; ++mt) {
    v16bf a = load_A_rm(xs + (w * 2 + mt) * 16 * 32, 32);
#pragma unroll
    for (int nt = 0; nt < 2; ++nt) {
      v8f acc = {};
      v16bf bb = load_frag_sw(wss + nt * 512);
      acc = wmma_bf16(a, bb, acc);
#pragma unroll
      for (int r = 0; r < 8; ++r) {
        int b = (w * 2 + mt) * 16 + r + hi * 8;
        int z = nt * 16 + n16;
        utmp[(((size_t)b * N1_ + k) * D1_ + z) * N0_ + j] = acc[r];
      }
    }
  }
}

// ---- K2: focus along last dim (64); one wave per row ----
__global__ void k2_focus(const float* __restrict__ u, bf16* __restrict__ ub) {
  const int w = threadIdx.x >> 5, l = threadIdx.x & 31;
  const long row = (long)blockIdx.x * 8 + w;          // < B*NI
  const float* p = u + row * 64;
  float a0 = (fmaxf(p[l], 0.f) + 1e-6f) * 0.2f;       // relu + eps, / SCALE(5)
  float a1 = (fmaxf(p[l + 32], 0.f) + 1e-6f) * 0.2f;
  float c0 = a0 * a0 * a0, c1 = a1 * a1 * a1;
  float s2 = a0 * a0 + a1 * a1;
  float s6 = c0 * c0 + c1 * c1;
#pragma unroll
  for (int m = 16; m >= 1; m >>= 1) {
    s2 += __shfl_xor(s2, m, 32);
    s6 += __shfl_xor(s6, m, 32);
  }
  float scale = sqrtf(s2) * rsqrtf(s6);               // ||u|| / ||u^3||
  bf16* q = ub + row * 64;
  q[l]      = (bf16)(c0 * scale);
  q[l + 32] = (bf16)(c1 * scale);
}

// ---- K3a: s[b,c] = sum_i u1[b,i,c] ----
__global__ void k3a_colsum(const bf16* __restrict__ u1, float* __restrict__ s) {
  const int b = blockIdx.x, c = threadIdx.x;          // 64 threads
  float acc = 0.f;
  const bf16* p = u1 + (long)b * NI_ * 64 + c;
  for (int i = 0; i < NI_; ++i) acc += (float)p[(long)i * 64];
  s[b * 64 + c] = acc;
}

// ---- K3b: z[b,i] = 1 / (dot(u1[b,i,:], s[b,:]) + eps) ----
__global__ void k3b_z(const bf16* __restrict__ u1, const float* __restrict__ s,
                      float* __restrict__ z) {
  const int w = threadIdx.x >> 5, l = threadIdx.x & 31;
  const long row = (long)blockIdx.x * 8 + w;          // b*NI + i
  const int b = (int)(row >> 11);
  const bf16* p = u1 + row * 64;
  const float* sb = s + b * 64;
  float d = (float)p[l] * sb[l] + (float)p[l + 32] * sb[l + 32];
#pragma unroll
  for (int m = 16; m >= 1; m >>= 1) d += __shfl_xor(d, m, 32);
  if (l == 0) z[row] = 1.f / (d + EPS_Z_);
}

// ---- K4: kv[b] = u2[b]^T @ u3[b]   (64x2048x64 per batch) ----
__global__ void k4_kv(const bf16* __restrict__ u2, const bf16* __restrict__ u3,
                      float* __restrict__ kv) {
  const int b = blockIdx.x;
  __shared__ alignas(16) bf16 s2s[4 * 512];   // 4 swizzled A tiles (transposed src)
  __shared__ alignas(16) bf16 s3s[4 * 512];   // 4 swizzled B tiles
  const int tid = threadIdx.x, w = tid >> 5, l = tid & 31, n16 = l & 15, hi = l >> 4;
  v8f acc[4] = {};
  const bf16* base2 = u2 + (long)b * NI_ * 64;
  const bf16* base3 = u3 + (long)b * NI_ * 64;
  for (int kc = 0; kc < NI_ / 32; ++kc) {
    for (int e = tid; e < 256; e += 128) {          // 8-bf16 chunks
      int i = e >> 3;                               // K row
      int c0 = (e & 7) << 3;                        // col base (one 16-block)
      v8bf v2 = *(const v8bf*)(base2 + (long)kc * 2048 + e * 8);
      v8bf v3 = *(const v8bf*)(base3 + (long)kc * 2048 + e * 8);
      int tile = c0 >> 4;
#pragma unroll
      for (int q = 0; q < 8; ++q) {
        int c = c0 + q;
        s2s[(tile << 9) + a_sw_idx(c & 15, i)] = v2[q]; // A[m=c][k=i]
        s3s[(tile << 9) + b_sw_idx(i, c & 15)] = v3[q]; // B[k=i][n=c]
      }
    }
    __syncthreads();
    v16bf a = load_frag_sw(s2s + w * 512);
#pragma unroll
    for (int nt = 0; nt < 4; ++nt) {
      v16bf bb = load_frag_sw(s3s + nt * 512);
      acc[nt] = wmma_bf16(a, bb, acc[nt]);
    }
    __syncthreads();
  }
#pragma unroll
  for (int nt = 0; nt < 4; ++nt)
#pragma unroll
    for (int r = 0; r < 8; ++r) {
      int c = w * 16 + r + hi * 8;
      int d = nt * 16 + n16;
      kv[((long)b * 64 + c) * 64 + d] = acc[nt][r];
    }
}

// ---- K5: x[b,i,:] = z[b,i] * (u1[b,i,:] @ kv[b])   (2048x64x64 per batch) ----
__global__ void k5_route(const bf16* __restrict__ u1, const float* __restrict__ kv,
                         const float* __restrict__ z, float* __restrict__ xout) {
  const int b = blockIdx.x, rb = blockIdx.y;          // rb < 16 -> 128 rows/block
  __shared__ alignas(16) bf16 kvs[8 * 512];           // (ks, ntile) swizzled B tiles
  __shared__ float zs[128];
  const int tid = threadIdx.x, w = tid >> 5, l = tid & 31, n16 = l & 15, hi = l >> 4;
  for (int e = tid; e < 1024; e += 256) {             // 4-f32 chunks
    float4 v = *(const float4*)(kv + (long)b * 4096 + e * 4);
    int kk = e >> 4;                                  // K dim 0..63
    int n0 = (e & 15) << 2;                           // within one 16-block
    int tile = ((kk >> 5) << 2) + (n0 >> 4);
    float vv[4] = {v.x, v.y, v.z, v.w};
#pragma unroll
    for (int q = 0; q < 4; ++q)
      kvs[(tile << 9) + b_sw_idx(kk & 31, (n0 + q) & 15)] = (bf16)vv[q];
  }
  if (tid < 128) zs[tid] = z[(long)b * NI_ + rb * 128 + tid];
  __syncthreads();
  const int i0 = rb * 128 + w * 16;
  v8f acc[4] = {};
#pragma unroll
  for (int ks = 0; ks < 2; ++ks) {
    v16bf a = load_A_rm(u1 + ((long)b * NI_ + i0) * 64 + ks * 32, 64);
#pragma unroll
    for (int nt = 0; nt < 4; ++nt) {
      v16bf bb = load_frag_sw(kvs + ((ks * 4 + nt) << 9));
      acc[nt] = wmma_bf16(a, bb, acc[nt]);
    }
  }
  float zv[8];
#pragma unroll
  for (int r = 0; r < 8; ++r) zv[r] = zs[w * 16 + r + hi * 8];
#pragma unroll
  for (int nt = 0; nt < 4; ++nt)
#pragma unroll
    for (int r = 0; r < 8; ++r) {
      int i = i0 + r + hi * 8;
      xout[((long)b * NI_ + i) * 64 + nt * 16 + n16] = acc[nt][r] * zv[r];
    }
}

// ---- K6a: window sums P[k][c*65+kk][b] (conv ⨯ time-sum collapse) + row sums Sx ----
__global__ void k6a_wins(const float* __restrict__ x, bf16* __restrict__ P,
                         float* __restrict__ Sx) {
  __shared__ float xr[8][64];
  const int w = threadIdx.x >> 5, l = threadIdx.x & 31;
  const long row = (long)blockIdx.x * 8 + w;          // b*NI + i
  const int b = (int)(row >> 11), i = (int)(row & 2047);
  const int k = i >> 5, c = i & 31;
  const float* p = x + row * 64;
  float v0 = p[l], v1 = p[l + 32];
  xr[w][l] = v0; xr[w][l + 32] = v1;
  __syncthreads();
  float tot = v0 + v1;
#pragma unroll
  for (int m = 16; m >= 1; m >>= 1) tot += __shfl_xor(tot, m, 32);
  if (l == 0) Sx[row] = tot;
  bf16* pp = P + ((long)k * KP_ + (long)c * KCONV) * 128 + b;
  // kk = l in [0,31]: window [0, kk+31]
  float sA = 0.f;
  for (int t = 0; t <= l + 31; ++t) sA += xr[w][t];
  pp[(long)l * 128] = (bf16)sA;
  // kk = l+32 in [32,63]: window [kk-32, 63]
  float sB = 0.f;
  for (int t = l; t < 64; ++t) sB += xr[w][t];
  pp[(long)(l + 32) * 128] = (bf16)sB;
  // kk = 64: window [32, 63]
  if (l == 0) {
    float sC = 0.f;
    for (int t = 32; t < 64; ++t) sC += xr[w][t];
    pp[(long)64 * 128] = (bf16)sC;
  }
}

// ---- K6b: Sy[b, k*32+d1] = conv_w[k][32 x 2080] @ P[k][2080 x 128]  per capsule k ----
__global__ void k6b_convsum(const bf16* __restrict__ wb, const bf16* __restrict__ P,
                            float* __restrict__ Sy) {
  const int k = blockIdx.x;
  __shared__ alignas(16) bf16 As[32 * 32];            // row-major [d1][kk]
  __shared__ alignas(16) bf16 Bs[8 * 512];            // 8 swizzled B tiles
  const int tid = threadIdx.x, w = tid >> 5, l = tid & 31, n16 = l & 15, hi = l >> 4;
  v8f acc[2][2] = {};
  const bf16* Abase = wb + (long)k * 32 * KP_;
  const bf16* Bbase = P + (long)k * KP_ * 128;
  for (int kc = 0; kc < 65; ++kc) {
    if (kc + 1 < 65)
      __builtin_prefetch(Bbase + (long)((kc + 1) * 32) * 128 + tid * 32, 0, 1);
    {                                                 // A: b128 -> b128 copy
      int d1 = tid >> 2, kko = (tid & 3) << 3;
      v8bf v = *(const v8bf*)(Abase + (long)d1 * KP_ + kc * 32 + kko);
      *(v8bf*)(As + tid * 8) = v;
    }
    for (int e = tid; e < 512; e += 128) {            // 8-bf16 chunks
      v8bf v = *(const v8bf*)(Bbase + (long)(kc * 32) * 128 + e * 8);
      int r = e >> 4;                                 // K row
      int n0 = (e & 15) << 3;                         // within one 16-block
      int tile = n0 >> 4;
#pragma unroll
      for (int q = 0; q < 8; ++q)
        Bs[(tile << 9) + b_sw_idx(r, (n0 + q) & 15)] = v[q];
    }
    __syncthreads();
    v16bf a0 = load_A_rm(As, 32);
    v16bf a1 = load_A_rm(As + 16 * 32, 32);
#pragma unroll
    for (int jn = 0; jn < 2; ++jn) {
      v16bf bb = load_frag_sw(Bs + ((w * 2 + jn) << 9));
      acc[0][jn] = wmma_bf16(a0, bb, acc[0][jn]);
      acc[1][jn] = wmma_bf16(a1, bb, acc[1][jn]);
    }
    __syncthreads();
  }
#pragma unroll
  for (int mt = 0; mt < 2; ++mt)
#pragma unroll
    for (int jn = 0; jn < 2; ++jn)
#pragma unroll
      for (int r = 0; r < 8; ++r) {
        int d1 = mt * 16 + r + hi * 8;
        int bcol = (w * 2 + jn) * 16 + n16;
        Sy[(long)bcol * NI_ + k * 32 + d1] = acc[mt][jn][r];
      }
}

// ---- K7: v = Sx + Sy + 64*bias ; squash over D1 (one wave per (b,k)) ----
__global__ void k7_squash(const float* __restrict__ Sx, const float* __restrict__ Sy,
                          const float* __restrict__ cb, float* __restrict__ out) {
  const int w = threadIdx.x >> 5, l = threadIdx.x & 31;
  const int bk = blockIdx.x * 8 + w;                  // < B*N1
  const int b = bk >> 6, k = bk & 63;
  const int i = k * 32 + l;
  const long idx = (long)b * NI_ + i;
  float v = Sx[idx] + Sy[idx] + 64.f * cb[i];
  float n2 = v * v;
#pragma unroll
  for (int m = 16; m >= 1; m >>= 1) n2 += __shfl_xor(n2, m, 32);
  float nrm = sqrtf(n2);
  float coef = 1.f - 1.f / (expf(nrm) + 1e-20f);
  out[idx] = coef * v / (nrm + 1e-20f);
}

extern "C" void kernel_launch(void* const* d_in, const int* in_sizes, int n_in,
                              void* d_out, int out_size, void* d_ws, size_t ws_size,
                              hipStream_t stream) {
  const float* x1 = (const float*)d_in[0];
  const float* x2 = (const float*)d_in[1];
  const float* x3 = (const float*)d_in[2];
  const float* W  = (const float*)d_in[3];
  const float* cw = (const float*)d_in[4];
  const float* cb = (const float*)d_in[5];
  float* out = (float*)d_out;

  char* ws = (char*)d_ws;
  size_t off = 0;
  auto alloc = [&](size_t bytes) -> void* {
    void* p = ws + off;
    off = (off + bytes + 255) & ~(size_t)255;
    return p;
  };
  float* utmp = (float*)alloc(sizeof(float) * (size_t)B_ * NI_ * 64); // 64MB, reused as routed x
  bf16* u1b = (bf16*)alloc(2ull * B_ * NI_ * 64);
  bf16* u2b = (bf16*)alloc(2ull * B_ * NI_ * 64);
  bf16* u3b = (bf16*)alloc(2ull * B_ * NI_ * 64);
  bf16* Pw  = (bf16*)alloc(2ull * N1_ * KP_ * 128);
  float* kv = (float*)alloc(4ull * B_ * 64 * 64);
  float* sC = (float*)alloc(4ull * B_ * 64);
  float* zB = (float*)alloc(4ull * B_ * NI_);
  float* Sy = (float*)alloc(4ull * B_ * NI_);
  float* Sx = (float*)alloc(4ull * B_ * NI_);
  bf16* cwb = (bf16*)alloc(2ull * NI_ * D1_ * KCONV);

  const int ncw8 = (NI_ * D1_ * KCONV) / 8;           // divisible: 4259840/8
  k0_cvt8<<<(ncw8 + 255) / 256, 256, 0, stream>>>(cw, cwb, ncw8);

  const float* xin[3] = {x1, x2, x3};
  bf16* ub[3] = {u1b, u2b, u3b};
  for (int t = 0; t < 3; ++t) {
    k1_transform<<<dim3(N0_, N1_), 128, 0, stream>>>(xin[t], W, utmp);
    k2_focus<<<(B_ * NI_) / 8, 256, 0, stream>>>(utmp, ub[t]);
  }

  k3a_colsum<<<B_, 64, 0, stream>>>(u1b, sC);
  k3b_z<<<(B_ * NI_) / 8, 256, 0, stream>>>(u1b, sC, zB);

  k4_kv<<<B_, 128, 0, stream>>>(u2b, u3b, kv);
  k5_route<<<dim3(B_, 16), 256, 0, stream>>>(u1b, kv, zB, utmp);

  k6a_wins<<<(B_ * NI_) / 8, 256, 0, stream>>>(utmp, Pw, Sx);
  k6b_convsum<<<N1_, 128, 0, stream>>>(cwb, Pw, Sy);

  k7_squash<<<(B_ * N1_) / 8, 256, 0, stream>>>(Sx, Sy, cb, out);
}
